// PrototypeCrossDomainTransfer_45947560132954
// MI455X (gfx1250) — matrix operand order repaired and checked by reading, not compiled, not run on hardware
//
#include <hip/hip_runtime.h>
#include <hip/hip_bf16.h>
#include <math.h>

typedef __attribute__((ext_vector_type(16))) __bf16 bf16x16;
typedef __attribute__((ext_vector_type(8)))  float  f32x8;

#define EPS_COS 1e-8f
#define EPS_LN  1e-5f
#define NEG_BIG (-1e30f)

// Branchless sorted-top3 insert: values via max/med3 pattern (v_max/v_min),
// indices via compare + select (v_cndmask). No EXEC divergence -> co-executes
// with XDL WMMA ops.
__device__ __forceinline__ void top3_insert(float v, int idx,
    float& v0, float& v1, float& v2, int& i0, int& i1, int& i2) {
  bool gt0 = v > v0;
  bool gt1 = v > v1;
  bool gt2 = v > v2;
  float nv0 = fmaxf(v, v0);
  float nv1 = fminf(fmaxf(v, v1), v0);   // med3(v, v0, v1)
  float nv2 = fminf(fmaxf(v, v2), v1);   // med3(v, v1, v2)
  int ni0 = gt0 ? idx : i0;
  int ni1 = gt0 ? i0 : (gt1 ? idx : i1);
  int ni2 = gt1 ? i1 : (gt2 ? idx : i2);
  v0 = nv0; v1 = nv1; v2 = nv2;
  i0 = ni0; i1 = ni1; i2 = ni2;
}

// ---------------------------------------------------------------------------
// Kernel 1: centers = mean_T(source_protos);  store bf16 centers + 1/||c||.
// One wave per proto, lane handles 2 of the 64 dims.
// ---------------------------------------------------------------------------
__global__ __launch_bounds__(256) void centers_kernel(
    const float* __restrict__ sp, __bf16* __restrict__ cbf,
    float* __restrict__ inv_norm, int P) {
  int wave = threadIdx.x >> 5;
  int lane = threadIdx.x & 31;
  int p = blockIdx.x * 8 + wave;
  if (p >= P) return;
  const float* base = sp + (size_t)p * 256;   // [T=4][D=64]
  float c0 = 0.f, c1 = 0.f;
  #pragma unroll
  for (int t = 0; t < 4; ++t) {
    float2 v = *(const float2*)(base + t * 64 + lane * 2);
    c0 += v.x; c1 += v.y;
  }
  c0 *= 0.25f; c1 *= 0.25f;
  float ss = c0 * c0 + c1 * c1;
  #pragma unroll
  for (int m = 16; m > 0; m >>= 1) ss += __shfl_xor(ss, m, 32);
  float inv = 1.0f / fmaxf(sqrtf(ss), EPS_COS);
  __bf16* outp = cbf + (size_t)p * 64 + lane * 2;
  outp[0] = (__bf16)c0;
  outp[1] = (__bf16)c1;
  if (lane == 0) inv_norm[p] = inv;
}

// ---------------------------------------------------------------------------
// Kernel 2: normalize target rows, pre-swizzle into WMMA A-fragment layout.
// tnA[tile][frag(0..1)][lane(0..31)][slot(0..15)] bf16,
// slot->K mapping k = f*32 + 16*(lane/16) + slot (contiguous order, matching B).
// ---------------------------------------------------------------------------
__global__ __launch_bounds__(32) void pack_tn_kernel(
    const float* __restrict__ tgt, __bf16* __restrict__ tnA) {
  __shared__ float row[16][64];
  __shared__ float inv[16];
  int tile = blockIdx.x, lane = threadIdx.x;
  for (int i = lane; i < 16 * 64; i += 32)
    row[i >> 6][i & 63] = tgt[(size_t)tile * 16 * 64 + i];
  __syncthreads();
  if (lane < 16) {
    float ss = 0.f;
    for (int d = 0; d < 64; ++d) ss += row[lane][d] * row[lane][d];
    inv[lane] = 1.0f / fmaxf(sqrtf(ss), EPS_COS);
  }
  __syncthreads();
  int m  = lane & 15;
  int hi = (lane >> 4) * 16;    // lanes 16-31 hold the upper 16 K of each frag
  #pragma unroll
  for (int f = 0; f < 2; ++f) {
    #pragma unroll
    for (int j = 0; j < 16; ++j) {
      int k = f * 32 + hi + j;
      tnA[(((size_t)tile * 2 + f) * 32 + lane) * 16 + j] =
          (__bf16)(row[m][k] * inv[m]);
    }
  }
}

// ---------------------------------------------------------------------------
// Kernel 3: fused sims matmul (bf16 WMMA, f32 acc) + per-row running top-3.
// Grid: (B/16 M-tiles) x (proto chunks).  One wave per block.
// ---------------------------------------------------------------------------
__global__ __launch_bounds__(32) void sims_topk_kernel(
    const __bf16* __restrict__ tnA, const __bf16* __restrict__ cbf,
    const float* __restrict__ inv_norm,
    float* __restrict__ cand_val, int* __restrict__ cand_idx,
    int PT, int nblk, int chunk) {
  int mtile = blockIdx.x;
  int pblk  = blockIdx.y;
  int lane  = threadIdx.x;

  const bf16x16* tnAv = (const bf16x16*)tnA;
  bf16x16 a0 = tnAv[((size_t)mtile * 2 + 0) * 32 + lane];   // K 0..31
  bf16x16 a1 = tnAv[((size_t)mtile * 2 + 1) * 32 + lane];   // K 32..63

  float tv[8][3];
  int   ti[8][3];
  #pragma unroll
  for (int r = 0; r < 8; ++r) {
    tv[r][0] = tv[r][1] = tv[r][2] = NEG_BIG;
    ti[r][0] = ti[r][1] = ti[r][2] = -1;
  }

  int tile0 = pblk * chunk;
  int ntile = PT - tile0; if (ntile > chunk) ntile = chunk;
  int col   = lane & 15;
  int khalf = (lane >> 4) * 16;   // lanes 16-31 hold the upper 16 K of a frag

  for (int t = 0; t < ntile; ++t) {
    int p = (tile0 + t) * 16 + col;
    const __bf16* bp = cbf + (size_t)p * 64;
    bf16x16 b0 = *(const bf16x16*)(bp + khalf);        // K 0..31 fragment
    bf16x16 b1 = *(const bf16x16*)(bp + 32 + khalf);   // K 32..63 fragment
    f32x8 c = {};
    c = __builtin_amdgcn_wmma_f32_16x16x32_bf16(false, a0, false, b0,
                                                (short)0, c, false, false);
    c = __builtin_amdgcn_wmma_f32_16x16x32_bf16(false, a1, false, b1,
                                                (short)0, c, false, false);
    float s = inv_norm[p];
    #pragma unroll
    for (int r = 0; r < 8; ++r) {
      float v = c[r] * s;  // lane<16: row=r ; lane>=16: row=8+r, col=lane&15
      top3_insert(v, p, tv[r][0], tv[r][1], tv[r][2],
                        ti[r][0], ti[r][1], ti[r][2]);
    }
  }

  // Cross-lane merge: 16 lanes of each half hold candidates for 8 rows.
  __shared__ float sval[32 * 8 * 3];
  __shared__ int   sidx[32 * 8 * 3];
  #pragma unroll
  for (int r = 0; r < 8; ++r)
    #pragma unroll
    for (int s2 = 0; s2 < 3; ++s2) {
      sval[(lane * 8 + r) * 3 + s2] = tv[r][s2];
      sidx[(lane * 8 + r) * 3 + s2] = ti[r][s2];
    }
  __syncthreads();
  if (lane < 16) {
    int m = lane;
    int lbase = (m < 8) ? 0 : 16;
    int r = (m < 8) ? m : m - 8;
    float v0 = NEG_BIG, v1 = NEG_BIG, v2 = NEG_BIG;
    int i0 = -1, i1 = -1, i2 = -1;
    for (int L = lbase; L < lbase + 16; ++L)
      for (int s2 = 0; s2 < 3; ++s2)
        top3_insert(sval[(L * 8 + r) * 3 + s2], sidx[(L * 8 + r) * 3 + s2],
                    v0, v1, v2, i0, i1, i2);
    int brow = mtile * 16 + m;
    size_t base = ((size_t)brow * nblk + pblk) * 3;
    cand_val[base + 0] = v0; cand_val[base + 1] = v1; cand_val[base + 2] = v2;
    cand_idx[base + 0] = i0; cand_idx[base + 1] = i1; cand_idx[base + 2] = i2;
  }
}

// ---------------------------------------------------------------------------
// Kernel 4: per batch row: global top-3 merge, softmax, gather (recompute
// centers from source_protos), concat, MLP with LayerNorm.
// ---------------------------------------------------------------------------
__global__ __launch_bounds__(256) void final_kernel(
    const float* __restrict__ cand_val, const int* __restrict__ cand_idx,
    int nblk, const float* __restrict__ sp, const float* __restrict__ tgt,
    const float* __restrict__ W1, const float* __restrict__ b1,
    const float* __restrict__ ln_g, const float* __restrict__ ln_b,
    const float* __restrict__ W2, const float* __restrict__ b2,
    const float* __restrict__ W3, const float* __restrict__ b3,
    float* __restrict__ out) {
  int b = blockIdx.x;
  int t = threadIdx.x;
  __shared__ float sv[256 * 3];
  __shared__ int   si[256 * 3];
  __shared__ float topv[3];
  __shared__ int   topi[3];
  __shared__ float w[3];
  __shared__ float x[192];
  __shared__ float h1[256];
  __shared__ float h2s[256];
  __shared__ float red[256], red2[256];

  // Phase A: merge nblk*3 candidates -> global top-3.
  int n = nblk * 3;
  float v0 = NEG_BIG, v1 = NEG_BIG, v2 = NEG_BIG;
  int i0 = -1, i1 = -1, i2 = -1;
  for (int c = t; c < n; c += 256)
    top3_insert(cand_val[(size_t)b * n + c], cand_idx[(size_t)b * n + c],
                v0, v1, v2, i0, i1, i2);
  sv[t * 3 + 0] = v0; sv[t * 3 + 1] = v1; sv[t * 3 + 2] = v2;
  si[t * 3 + 0] = i0; si[t * 3 + 1] = i1; si[t * 3 + 2] = i2;
  __syncthreads();
  if (t == 0) {
    float a0 = NEG_BIG, a1 = NEG_BIG, a2 = NEG_BIG;
    int   j0 = -1, j1 = -1, j2 = -1;
    for (int c = 0; c < 256 * 3; ++c)
      top3_insert(sv[c], si[c], a0, a1, a2, j0, j1, j2);
    topv[0] = a0; topv[1] = a1; topv[2] = a2;
    topi[0] = j0; topi[1] = j1; topi[2] = j2;
    float e0 = expf(a0 - a0), e1 = expf(a1 - a0), e2 = expf(a2 - a0);
    float den = e0 + e1 + e2;
    w[0] = e0 / den; w[1] = e1 / den; w[2] = e2 / den;
  }
  __syncthreads();

  // Phase B: gather centers (recompute mean over T) and build concat x[192].
  if (t < 64) {
    float cz[3];
    #pragma unroll
    for (int k = 0; k < 3; ++k) {
      const float* pp = sp + (size_t)topi[k] * 256 + t;
      cz[k] = 0.25f * (pp[0] + pp[64] + pp[128] + pp[192]);
    }
    x[t]       = w[0] * cz[0] + w[1] * cz[1] + w[2] * cz[2];
    x[64 + t]  = cz[0];
    x[128 + t] = tgt[(size_t)b * 64 + t];
  }
  __syncthreads();

  // Phase C: h = LN(x @ W1 + b1) -> relu.
  float acc = b1[t];
  for (int i = 0; i < 192; ++i) acc += x[i] * W1[(size_t)i * 256 + t];
  red[t] = acc; red2[t] = acc * acc;
  __syncthreads();
  for (int s = 128; s > 0; s >>= 1) {
    if (t < s) { red[t] += red[t + s]; red2[t] += red2[t + s]; }
    __syncthreads();
  }
  float mu  = red[0] * (1.0f / 256.0f);
  float var = red2[0] * (1.0f / 256.0f) - mu * mu;
  float hn  = (acc - mu) * rsqrtf(var + EPS_LN) * ln_g[t] + ln_b[t];
  h1[t] = fmaxf(hn, 0.f);
  __syncthreads();

  // Phase D: h2 = relu(h1 @ W2 + b2).
  float a2 = b2[t];
  for (int i = 0; i < 256; ++i) a2 += h1[i] * W2[(size_t)i * 256 + t];
  h2s[t] = fmaxf(a2, 0.f);
  __syncthreads();

  // Phase E: out = h2 @ W3 + b3.
  if (t < 64) {
    float o = b3[t];
    for (int i = 0; i < 256; ++i) o += h2s[i] * W3[(size_t)i * 64 + t];
    out[(size_t)b * 64 + t] = o;
  }
}

// ---------------------------------------------------------------------------
extern "C" void kernel_launch(void* const* d_in, const int* in_sizes, int n_in,
                              void* d_out, int out_size, void* d_ws, size_t ws_size,
                              hipStream_t stream) {
  const float* target = (const float*)d_in[0];
  const float* sp     = (const float*)d_in[1];
  const float* W1     = (const float*)d_in[2];
  const float* b1     = (const float*)d_in[3];
  const float* ln_g   = (const float*)d_in[4];
  const float* ln_b   = (const float*)d_in[5];
  const float* W2     = (const float*)d_in[6];
  const float* b2     = (const float*)d_in[7];
  const float* W3     = (const float*)d_in[8];
  const float* b3     = (const float*)d_in[9];

  const int B = 512, P = 200000;
  const int MT = B / 16;            // 32 M-tiles
  const int PT = P / 16;            // 12500 proto tiles
  const int CHUNK = 128;            // proto tiles per workgroup (2048 protos)
  const int nblk = (PT + CHUNK - 1) / CHUNK;   // 98

  char* ws = (char*)d_ws;
  size_t off = 0;
  auto take = [&](size_t bytes) -> char* {
    char* p = ws + off;
    off = (off + bytes + 255) & ~(size_t)255;
    return p;
  };
  __bf16* cbf      = (__bf16*)take((size_t)P * 64 * 2);
  float*  inv_norm = (float*) take((size_t)P * 4);
  __bf16* tnA      = (__bf16*)take((size_t)MT * 2 * 32 * 16 * 2);
  float*  cand_val = (float*) take((size_t)B * nblk * 3 * 4);
  int*    cand_idx = (int*)   take((size_t)B * nblk * 3 * 4);

  centers_kernel<<<(P + 7) / 8, 256, 0, stream>>>(sp, cbf, inv_norm, P);
  pack_tn_kernel<<<MT, 32, 0, stream>>>(target, tnA);
  sims_topk_kernel<<<dim3(MT, nblk), 32, 0, stream>>>(
      tnA, cbf, inv_norm, cand_val, cand_idx, PT, nblk, CHUNK);
  final_kernel<<<B, 256, 0, stream>>>(cand_val, cand_idx, nblk, sp, target,
                                      W1, b1, ln_g, ln_b, W2, b2, W3, b3,
                                      (float*)d_out);
}
